// RGCNLabelEncoder_35158602285585
// MI455X (gfx1250) — matrix operand order repaired and checked by reading, not compiled, not run on hardware
//
#include <hip/hip_runtime.h>
#include <hip/hip_bf16.h>

typedef __attribute__((ext_vector_type(16))) __bf16 v16bf;
typedef __attribute__((ext_vector_type(8)))  __bf16 v8bf;
typedef __attribute__((ext_vector_type(8)))  float  v8f;

#define N_NODES 50000
#define N_EDGES 400000
#define N_REL   4
#define N_BLK   4

// ---------------------------------------------------------------- helpers

__device__ __forceinline__ unsigned short f32_to_bf16_rne(float f) {
  union { float f; unsigned u; } v; v.f = f;
  unsigned u = v.u;
  if ((u & 0x7fffffffu) > 0x7f800000u) return (unsigned short)((u >> 16) | 0x40); // NaN
  u += 0x7fffu + ((u >> 16) & 1u);                                               // RNE
  return (unsigned short)(u >> 16);
}

__device__ __forceinline__ void atomic_add_f32(float* p, float v) {
  unsafeAtomicAdd(p, v);   // hardware global_atomic_add_f32
}

// A fragment: lane(0..15)=row M, halves give K octets {h*8..h*8+7, 16+h*8..}
// B fragment: lane(0..15)=col N, halves give K octets {h*8..h*8+7, 16+h*8..}
__device__ __forceinline__ v8f wmma_bf16(v8bf alo, v8bf ahi, v8bf blo, v8bf bhi, v8f c) {
  v16bf a = __builtin_shufflevector(alo, ahi, 0,1,2,3,4,5,6,7,8,9,10,11,12,13,14,15);
  v16bf b = __builtin_shufflevector(blo, bhi, 0,1,2,3,4,5,6,7,8,9,10,11,12,13,14,15);
  return __builtin_amdgcn_wmma_f32_16x16x32_bf16(false, a, false, b, (short)0, c, false, false);
}

// ---------------------------------------------------------------- sort by relation

__global__ void zero_meta_k(int* meta) { if (threadIdx.x < 64) meta[threadIdx.x] = 0; }

__global__ void hist_k(const int* __restrict__ etype, int* __restrict__ meta) {
  int e = blockIdx.x * blockDim.x + threadIdx.x;
  if (e < N_EDGES) atomicAdd(&meta[etype[e]], 1);
}

__global__ void scan_k(int* meta) {
  if (blockIdx.x == 0 && threadIdx.x == 0) {
    int s = 0;
    for (int r = 0; r < N_REL; ++r) { meta[8 + r] = s; meta[4 + r] = s; s += meta[r]; }
  }
}

__global__ void scatter_k(const int* __restrict__ etype, int* __restrict__ meta,
                          int* __restrict__ sorted) {
  int e = blockIdx.x * blockDim.x + threadIdx.x;
  if (e < N_EDGES) {
    int p = atomicAdd(&meta[4 + etype[e]], 1);
    sorted[p] = e;
  }
}

// ---------------------------------------------------------------- bf16 prep

// f32 [rows][cols] -> bf16 [rows][colspad], zero padded K
__global__ void cvt_pad_rows_k(const float* __restrict__ src, unsigned short* __restrict__ dst,
                               int rows, int cols, int colspad) {
  long long i = (long long)blockIdx.x * blockDim.x + threadIdx.x;
  if (i >= (long long)rows * colspad) return;
  int r = (int)(i / colspad), c = (int)(i - (long long)r * colspad);
  float v = (c < cols) ? src[(long long)r * cols + c] : 0.f;
  dst[i] = f32_to_bf16_rne(v);
}

// nmat matrices f32 [si][so] -> bf16 transposed padded [sopad][sipad]
__global__ void cvt_tr_pad_k(const float* __restrict__ src, unsigned short* __restrict__ dst,
                             int nmat, int si, int so, int sipad, int sopad) {
  long long i = (long long)blockIdx.x * blockDim.x + threadIdx.x;
  long long per = (long long)sopad * sipad;
  if (i >= (long long)nmat * per) return;
  int mat = (int)(i / per);
  long long rem = i - (long long)mat * per;
  int n = (int)(rem / sipad), k = (int)(rem - (long long)n * sipad);
  float v = (n < so && k < si) ? src[((long long)mat * si + k) * so + n] : 0.f;
  dst[i] = f32_to_bf16_rne(v);
}

// leaky_relu(acc [rows][cols]) -> bf16 [rows][colspad]
__global__ void finalize_act_k(const float* __restrict__ acc, unsigned short* __restrict__ dst,
                               int rows, int cols, int colspad) {
  long long i = (long long)blockIdx.x * blockDim.x + threadIdx.x;
  if (i >= (long long)rows * colspad) return;
  int r = (int)(i / colspad), c = (int)(i - (long long)r * colspad);
  float v = 0.f;
  if (c < cols) { float x = acc[(long long)r * cols + c]; v = x > 0.f ? x : 0.2f * x; }
  dst[i] = f32_to_bf16_rne(v);
}

// ---------------------------------------------------------------- self-loop GEMM (WMMA)
// C[row][col] = A(bf16,[rows][kpad]) @ Bt(bf16,[ncols][kpad])^T + bias ; non-atomic init of acc
__global__ void selfloop_gemm(const unsigned short* __restrict__ A, int kpad,
                              const unsigned short* __restrict__ Bt,
                              const float* __restrict__ bias,
                              float* __restrict__ C, int ldc, int ncols) {
  const int lane = threadIdx.x & 31;
  const int wave = threadIdx.x >> 5;
  const int ntile = blockIdx.y * 8 + wave;
  if (ntile * 16 >= ncols) return;                    // wave-uniform
  const int row0 = blockIdx.x * 16;
  const int col0 = ntile * 16;
  const int half = lane >> 4;
  const int l15  = lane & 15;
  const unsigned short* ap = A  + (size_t)(row0 + l15) * kpad + half * 8;
  const unsigned short* bp = Bt + (size_t)(col0 + l15) * kpad + half * 8;
  v8f acc = {};
  for (int kc = 0; kc < kpad; kc += 32) {
    v8bf alo = *(const v8bf*)(ap + kc);
    v8bf ahi = *(const v8bf*)(ap + kc + 16);
    v8bf blo = *(const v8bf*)(bp + kc);
    v8bf bhi = *(const v8bf*)(bp + kc + 16);
    acc = wmma_bf16(alo, ahi, blo, bhi, acc);
  }
  float bs = bias[col0 + l15];
#pragma unroll
  for (int v = 0; v < 8; ++v) {
    int m = v + half * 8;                             // D layout: VGPR v -> M = v + 8*half
    C[(size_t)(row0 + m) * ldc + col0 + l15] = acc[v] + bs;
  }
}

// ---------------------------------------------------------------- edge messages (WMMA + scatter)
// One block = 16 edges of one relation. Gather 16 src rows (bf16, zero-padded per diag
// block to sipad) into LDS, run N_BLK*ntc WMMA tile jobs over 8 waves, atomically
// scatter D rows into acc[dst].
__global__ void edge_msg(const unsigned short* __restrict__ Hbf, int lda,
                         const unsigned short* __restrict__ Wt,     // [R][NB][sopad][sipad]
                         const int* __restrict__ meta,
                         const int* __restrict__ sorted,
                         const int* __restrict__ src,
                         const int* __restrict__ dst,
                         float* __restrict__ acc, int ldc,
                         int si, int so, int sipad, int ntc) {
  __shared__ __align__(16) unsigned short s_h[16 * N_BLK * 128];    // max 16 KB
  __shared__ int s_src[16], s_dst[16], s_valid[16];

  // resolve (relation, tile) from per-relation counts
  int t = blockIdx.x;
  int rel = -1, tile = 0, cnt = 0, start = 0;
  for (int r = 0; r < N_REL; ++r) {
    int c = meta[r];
    int tr = (c + 15) >> 4;
    if (t < tr) { rel = r; tile = t; cnt = c; start = meta[8 + r]; break; }
    t -= tr;
  }
  if (rel < 0) return;                                // block-uniform

  const int tid = threadIdx.x;
  if (tid < 16) {
    int idx = tile * 16 + tid;
    int valid = idx < cnt;
    int e = valid ? sorted[start + idx] : 0;
    s_valid[tid] = valid;
    s_src[tid] = valid ? src[e] : 0;
    s_dst[tid] = valid ? dst[e] : 0;
  }
  __syncthreads();

  const int rowstride = N_BLK * sipad;
  for (int idx = tid; idx < 16 * rowstride; idx += blockDim.x) {
    int m = idx / rowstride;
    int j = idx - m * rowstride;
    int b = j / sipad;
    int k = j - b * sipad;
    unsigned short v = 0;
    if (s_valid[m] && k < si) v = Hbf[(size_t)s_src[m] * lda + b * si + k];
    s_h[idx] = v;
  }
  __syncthreads();

  const int lane = tid & 31;
  const int wave = tid >> 5;
  const int half = lane >> 4;
  const int l15  = lane & 15;
  const int sopad = ntc * 16;
  const int njobs = N_BLK * ntc;

  for (int job = wave; job < njobs; job += 8) {
    int b  = job / ntc;
    int nc = job - b * ntc;
    const unsigned short* ap = &s_h[l15 * rowstride + b * sipad + half * 8];
    const unsigned short* bp =
        Wt + ((size_t)((rel * N_BLK + b) * sopad + nc * 16 + l15)) * sipad + half * 8;
    v8f c = {};
    for (int kc = 0; kc < sipad; kc += 32) {
      v8bf alo = *(const v8bf*)(ap + kc);
      v8bf ahi = *(const v8bf*)(ap + kc + 16);
      v8bf blo = *(const v8bf*)(bp + kc);
      v8bf bhi = *(const v8bf*)(bp + kc + 16);
      c = wmma_bf16(alo, ahi, blo, bhi, c);
    }
    int ncol = nc * 16 + l15;
    if (ncol < so) {
#pragma unroll
      for (int v = 0; v < 8; ++v) {
        int m = v + half * 8;
        if (s_valid[m])
          atomic_add_f32(&acc[(size_t)s_dst[m] * ldc + b * so + ncol], c[v]);
      }
    }
  }
}

// ---------------------------------------------------------------- launch

extern "C" void kernel_launch(void* const* d_in, const int* in_sizes, int n_in,
                              void* d_out, int out_size, void* d_ws, size_t ws_size,
                              hipStream_t stream) {
  const float* init_emb = (const float*)d_in[0];
  const float* W1    = (const float*)d_in[1];
  const float* loop1 = (const float*)d_in[2];
  const float* b1    = (const float*)d_in[3];
  const float* W2    = (const float*)d_in[4];
  const float* loop2 = (const float*)d_in[5];
  const float* b2    = (const float*)d_in[6];
  const int* src   = (const int*)d_in[7];
  const int* dst   = (const int*)d_in[8];
  const int* etype = (const int*)d_in[9];
  float* out = (float*)d_out;

  char* ws = (char*)d_ws;
  size_t off = 0;
  auto take = [&](size_t bytes) -> char* {
    char* p = ws + off; off = (off + bytes + 255) & ~(size_t)255; return p;
  };
  int* meta              = (int*)take(64 * sizeof(int));
  int* sorted            = (int*)take((size_t)N_EDGES * 4);
  unsigned short* h0bf   = (unsigned short*)take((size_t)N_NODES * 320 * 2);
  unsigned short* hmidbf = (unsigned short*)take((size_t)N_NODES * 416 * 2);
  unsigned short* loop1t = (unsigned short*)take((size_t)400 * 320 * 2);
  unsigned short* loop2t = (unsigned short*)take((size_t)512 * 416 * 2);
  unsigned short* W1t    = (unsigned short*)take((size_t)16 * 112 * 96 * 2);
  unsigned short* W2t    = (unsigned short*)take((size_t)16 * 128 * 128 * 2);

  float* acc1 = out;   // layer-1 accumulator aliases d_out [N][400]; consumed before layer 2

  // edge sort by relation
  zero_meta_k<<<1, 64, 0, stream>>>(meta);
  hist_k<<<(N_EDGES + 255) / 256, 256, 0, stream>>>(etype, meta);
  scan_k<<<1, 1, 0, stream>>>(meta);
  scatter_k<<<(N_EDGES + 255) / 256, 256, 0, stream>>>(etype, meta, sorted);

  // bf16 prep (K padded to multiples of 32; weights transposed for contiguous B frags)
  {
    long long tot = (long long)N_NODES * 320;
    cvt_pad_rows_k<<<(unsigned)((tot + 255) / 256), 256, 0, stream>>>(init_emb, h0bf, N_NODES, 300, 320);
  }
  cvt_tr_pad_k<<<(400 * 320 + 255) / 256, 256, 0, stream>>>(loop1, loop1t, 1, 300, 400, 320, 400);
  cvt_tr_pad_k<<<(512 * 416 + 255) / 256, 256, 0, stream>>>(loop2, loop2t, 1, 400, 512, 416, 512);
  cvt_tr_pad_k<<<(16 * 112 * 96 + 255) / 256, 256, 0, stream>>>(W1, W1t, 16, 75, 100, 96, 112);
  cvt_tr_pad_k<<<(16 * 128 * 128 + 255) / 256, 256, 0, stream>>>(W2, W2t, 16, 100, 128, 128, 128);

  // ---- layer 1: acc1 = h0 @ loop1 + b1, then scatter edge messages
  {
    dim3 g(N_NODES / 16, 4);   // 25 col tiles over 8 waves -> 4 groups (tail waves idle)
    selfloop_gemm<<<g, 256, 0, stream>>>(h0bf, 320, loop1t, b1, acc1, 400, 400);
  }
  edge_msg<<<N_EDGES / 16 + N_REL, 256, 0, stream>>>(h0bf, 320, W1t, meta, sorted,
                                                     src, dst, acc1, 400, 75, 100, 96, 7);
  {
    long long tot = (long long)N_NODES * 416;
    finalize_act_k<<<(unsigned)((tot + 255) / 256), 256, 0, stream>>>(acc1, hmidbf, N_NODES, 400, 416);
  }

  // ---- layer 2: out = h_mid @ loop2 + b2, then scatter edge messages
  {
    dim3 g(N_NODES / 16, 4);   // 32 col tiles over 8 waves -> 4 groups exactly
    selfloop_gemm<<<g, 256, 0, stream>>>(hmidbf, 416, loop2t, b2, out, 512, 512);
  }
  edge_msg<<<N_EDGES / 16 + N_REL, 256, 0, stream>>>(hmidbf, 416, W2t, meta, sorted,
                                                     src, dst, out, 512, 100, 128, 128, 8);
}
